// DDPG_4380866642504
// MI455X (gfx1250) — matrix operand order repaired and checked by reading, not compiled, not run on hardware
//
#include <hip/hip_runtime.h>
#include <hip/hip_bf16.h>

// ---------------- problem constants (from reference) ----------------
#define B_      1024
#define S_      256
#define D_      128
#define N_      100000            // divisible by 32
#define K_      10
#define IPC     1024              // items per chunk
#define NCHUNK  ((N_ + IPC - 1) / IPC)   // 98 (last chunk = 672 items, %32==0)
#define WAVES   8
#define ROWS_PER_BLOCK 128        // 8 waves * 16 rows

// ---------------- vector types (trivial, union-safe) ----------------
typedef __bf16        v16bf __attribute__((ext_vector_type(16)));
typedef __bf16        v8bf  __attribute__((ext_vector_type(8)));
typedef float         v8f   __attribute__((ext_vector_type(8)));
typedef float         f32x4 __attribute__((ext_vector_type(4)));
typedef unsigned int  u32x4 __attribute__((ext_vector_type(4)));
typedef int           i32x4 __attribute__((ext_vector_type(4)));

union AFrag { v16bf v; v8bf h[2]; };
union AccU  { v8f   v; f32x4 f[2]; };

#define LDS_AS __attribute__((address_space(3)))
#define GLB_AS __attribute__((address_space(1)))

#if defined(__has_builtin)
#  if __has_builtin(__builtin_amdgcn_global_load_async_to_lds_b128)
#    define HAVE_ASYNC_B128 1
#  endif
#  if __has_builtin(__builtin_amdgcn_s_wait_asynccnt)
#    define HAVE_WAIT_ASYNC 1
#  endif
#endif

// one 16-byte async global->LDS copy (CDNA5 async path, tracked by ASYNCcnt)
__device__ __forceinline__ void async_copy_b128(const unsigned short* g,
                                                unsigned short* l) {
#if defined(HAVE_ASYNC_B128)
  // builtin prototype (from probe diagnostic): param0 = AS1 int4*
  __builtin_amdgcn_global_load_async_to_lds_b128(
      (GLB_AS i32x4*)(size_t)g, (LDS_AS i32x4*)l, 0, 0);
#else
  const unsigned lds32 = (unsigned)(unsigned long long)l;  // LDS byte offset
  asm volatile("global_load_async_to_lds_b128 %0, %1, off"
               :: "v"(lds32), "v"((unsigned long long)g) : "memory");
#endif
}

__device__ __forceinline__ void wait_async_le2() {
#if defined(HAVE_WAIT_ASYNC)
  __builtin_amdgcn_s_wait_asynccnt(2);
#else
  asm volatile("s_wait_asynccnt 0x2" ::: "memory");
#endif
}
__device__ __forceinline__ void wait_async_0() {
#if defined(HAVE_WAIT_ASYNC)
  __builtin_amdgcn_s_wait_asynccnt(0);
#else
  asm volatile("s_wait_asynccnt 0x0" ::: "memory");
#endif
}

__device__ __forceinline__ unsigned short f32_to_bf16(float f) {
  unsigned int u = __float_as_uint(f);
  unsigned int r = 0x7FFFu + ((u >> 16) & 1u);   // round-to-nearest-even
  return (unsigned short)((u + r) >> 16);
}

// -------------------------------------------------------------------
// Kernel 1: action = normalize(state @ W + b), write bf16 rows.
// -------------------------------------------------------------------
__global__ void actor_norm_kernel(const float* __restrict__ state,
                                  const float* __restrict__ W,
                                  const float* __restrict__ b,
                                  unsigned short* __restrict__ actBf16) {
  __shared__ float srow[S_];
  __shared__ float red[D_];
  const int row = blockIdx.x;
  const int d   = threadIdx.x;                 // 0..127
  srow[d]       = state[row * S_ + d];
  srow[d + 128] = state[row * S_ + d + 128];
  __syncthreads();
  float acc = b[d];
#pragma unroll 8
  for (int s = 0; s < S_; ++s) acc = fmaf(srow[s], W[s * D_ + d], acc);
  red[d] = acc * acc;
  __syncthreads();
  for (int off = 64; off > 0; off >>= 1) {
    if (d < off) red[d] += red[d + off];
    __syncthreads();
  }
  const float inv = 1.0f / sqrtf(red[0]);
  actBf16[row * D_ + d] = f32_to_bf16(acc * inv);
}

// -------------------------------------------------------------------
// Kernel 2: normalize item rows, convert to bf16 (L2-resident 25.6MB).
// -------------------------------------------------------------------
__global__ void item_norm_kernel(const float* __restrict__ items,
                                 unsigned short* __restrict__ itemsBf16) {
  const int wave = threadIdx.x >> 5;
  const int lane = threadIdx.x & 31;
  const int row  = blockIdx.x * 8 + wave;
  const f32x4 v = *(const f32x4*)(items + row * D_ + lane * 4);
  float ss = v.x * v.x + v.y * v.y + v.z * v.z + v.w * v.w;
#pragma unroll
  for (int off = 16; off > 0; off >>= 1) ss += __shfl_xor(ss, off, 32);
  const float inv = 1.0f / sqrtf(ss);
  unsigned short* dst = itemsBf16 + row * D_ + lane * 4;
  dst[0] = f32_to_bf16(v.x * inv);
  dst[1] = f32_to_bf16(v.y * inv);
  dst[2] = f32_to_bf16(v.z * inv);
  dst[3] = f32_to_bf16(v.w * inv);
}

// -------------------------------------------------------------------
// Kernel 3: fused cosine-score GEMM (bf16 WMMA, f32 accum) + per-chunk
// top-10. grid = (98 chunks) x (8 row-tiles); block = 256 thr = 8 waves.
// Double-buffered async item tiles; two independent WMMA accumulator
// chains per tile (2 x 16 item columns).
// -------------------------------------------------------------------
__global__ void score_topk_kernel(const unsigned short* __restrict__ actBf16,
                                  const unsigned short* __restrict__ itemsBf16,
                                  float* __restrict__ pscore,
                                  int* __restrict__ pidx) {
  __shared__ __align__(16) unsigned short tiles[2][32 * D_]; // 2 x 8 KB
  __shared__ __align__(16) float sw[WAVES * 32 * 16];        // 16 KB scratch

  const int chunk     = blockIdx.x;
  const int chunkBase = chunk * IPC;
  const int nItems    = min(IPC, N_ - chunkBase);
  const int ntile     = nItems >> 5;                       // chunks are %32
  const int tid  = threadIdx.x;
  const int wave = tid >> 5, lane = tid & 31;
  const int m  = lane & 15;                                // row/col within 16
  const int hi = lane >> 4;                                // half-K selector
  const int rowBase = blockIdx.y * ROWS_PER_BLOCK + wave * 16;

  // A-fragments: 16 rows x 128 K, contiguous K order matching B:
  // lane (hi=0): K s*32+0..15; lane (hi=1): K s*32+16..31.
  AFrag afrag[4];
  {
    const unsigned short* ap = actBf16 + (rowBase + m) * D_;
#pragma unroll
    for (int s = 0; s < 4; ++s) {
      afrag[s].h[0] = *(const v8bf*)(ap + s * 32 + hi * 16);
      afrag[s].h[1] = *(const v8bf*)(ap + s * 32 + hi * 16 + 8);
    }
  }

  float ts[K_]; int ti[K_];
#pragma unroll
  for (int k = 0; k < K_; ++k) { ts[k] = -__builtin_inff(); ti[k] = 0; }

  float* mysw = sw + wave * 32 * 16;

  // each thread async-copies its 32B of the 8KB tile (2 x b128)
  auto issue_tile = [&](int t2, unsigned short* buf) {
    const unsigned short* g =
        itemsBf16 + (size_t)(chunkBase + t2 * 32) * D_ + tid * 16;
    unsigned short* l = buf + tid * 16;
    async_copy_b128(g, l);
    async_copy_b128(g + 8, l + 8);
  };

  issue_tile(0, tiles[0]);

  for (int t = 0; t < ntile; ++t) {
    __syncthreads();                       // all reads of tiles[(t+1)&1] done
    if (t + 1 < ntile) {
      issue_tile(t + 1, tiles[(t + 1) & 1]);
      wait_async_le2();                    // tile t landed; t+1 stays in flight
    } else {
      wait_async_0();
    }
    __syncthreads();                       // every wave's copies of tile t landed

    const unsigned short* cur = tiles[t & 1];

    AccU acc0, acc1;
#pragma unroll
    for (int i = 0; i < 2; ++i) {
      acc0.f[i] = (f32x4){0.f, 0.f, 0.f, 0.f};
      acc1.f[i] = (f32x4){0.f, 0.f, 0.f, 0.f};
    }
#pragma unroll
    for (int s = 0; s < 4; ++s) {          // K slices of 32; two indep chains
      // B layout: lanes 0-15 hold K 0..15 of column n, lanes 16-31 K 16..31
      const v16bf b0 = *(const v16bf*)(cur + (m)      * D_ + s * 32 + hi * 16);
      const v16bf b1 = *(const v16bf*)(cur + (16 + m) * D_ + s * 32 + hi * 16);
      acc0.v = __builtin_amdgcn_wmma_f32_16x16x32_bf16(
          false, afrag[s].v, false, b0, (short)0, acc0.v, false, false);
      acc1.v = __builtin_amdgcn_wmma_f32_16x16x32_bf16(
          false, afrag[s].v, false, b1, (short)0, acc1.v, false, false);
    }
    // dump 16x32 scores: lane L -> sw[L][0..15]; score(M, c*16+N) at
    // sw[N + 16*(M>=8)][c*8 + (M&7)]
    *(f32x4*)(mysw + lane * 16)      = acc0.f[0];
    *(f32x4*)(mysw + lane * 16 + 4)  = acc0.f[1];
    *(f32x4*)(mysw + lane * 16 + 8)  = acc1.f[0];
    *(f32x4*)(mysw + lane * 16 + 12) = acc1.f[1];
    asm volatile("s_wait_dscnt 0" ::: "memory");           // wave-local visibility

    if (lane < 16) {                       // lane m owns row rowBase+m
      const int r = m & 7, h2 = (m >> 3) << 4;
      const int ibase = chunkBase + t * 32;
#pragma unroll
      for (int j = 0; j < 32; ++j) {       // ascending item index (tie order)
        const int n = j & 15, cc = j >> 4;
        const float sc = mysw[(n + h2) * 16 + cc * 8 + r];
        if (sc > ts[K_ - 1]) {             // sorted-insert, static indices only
          const int id = ibase + j;
#pragma unroll
          for (int p = K_ - 1; p > 0; --p) {
            const bool shift = sc > ts[p - 1];
            const float ns = shift ? ts[p - 1] : sc;
            const int   ni = shift ? ti[p - 1] : id;
            if (sc > ts[p]) { ts[p] = ns; ti[p] = ni; }
          }
          if (sc > ts[0]) { ts[0] = sc; ti[0] = id; }
        }
      }
    }
  }

  if (lane < 16) {
    const int row = rowBase + m;
    float* ps = pscore + ((size_t)row * NCHUNK + chunk) * K_;
    int*   pi = pidx   + ((size_t)row * NCHUNK + chunk) * K_;
#pragma unroll
    for (int k = 0; k < K_; ++k) { ps[k] = ts[k]; pi[k] = ti[k]; }
  }
}

// -------------------------------------------------------------------
// Kernel 4: merge 98 partial top-10 lists per row -> final 10 indices.
// -------------------------------------------------------------------
__global__ void merge_topk_kernel(const float* __restrict__ pscore,
                                  const int* __restrict__ pidx,
                                  int* __restrict__ out) {
  __shared__ float ls[32 * K_];
  __shared__ int   li[32 * K_];
  const int row  = blockIdx.x;
  const int lane = threadIdx.x;
  const int ncand = NCHUNK * K_;                           // 980
  float ts[K_]; int ti[K_];
#pragma unroll
  for (int k = 0; k < K_; ++k) { ts[k] = -__builtin_inff(); ti[k] = 0; }
  const float* ps = pscore + (size_t)row * ncand;
  const int*   pi = pidx   + (size_t)row * ncand;
  for (int c = lane; c < ncand; c += 32) {
    const float sc = ps[c];
    if (sc > ts[K_ - 1]) {
      const int id = pi[c];
#pragma unroll
      for (int p = K_ - 1; p > 0; --p) {
        const bool shift = sc > ts[p - 1];
        const float ns = shift ? ts[p - 1] : sc;
        const int   ni = shift ? ti[p - 1] : id;
        if (sc > ts[p]) { ts[p] = ns; ti[p] = ni; }
      }
      if (sc > ts[0]) { ts[0] = sc; ti[0] = id; }
    }
  }
#pragma unroll
  for (int k = 0; k < K_; ++k) { ls[lane * K_ + k] = ts[k]; li[lane * K_ + k] = ti[k]; }
  __syncthreads();
  if (lane == 0) {
    float fs[K_]; int fi[K_];
#pragma unroll
    for (int k = 0; k < K_; ++k) { fs[k] = -__builtin_inff(); fi[k] = 0; }
    for (int j = 0; j < 32 * K_; ++j) {
      const float sc = ls[j];
      if (sc > fs[K_ - 1]) {
        const int id = li[j];
#pragma unroll
        for (int p = K_ - 1; p > 0; --p) {
          const bool shift = sc > fs[p - 1];
          const float ns = shift ? fs[p - 1] : sc;
          const int   ni = shift ? fi[p - 1] : id;
          if (sc > fs[p]) { fs[p] = ns; fi[p] = ni; }
        }
        if (sc > fs[0]) { fs[0] = sc; fi[0] = id; }
      }
    }
#pragma unroll
    for (int k = 0; k < K_; ++k) out[row * K_ + k] = fi[k];
  }
}

// -------------------------------------------------------------------
extern "C" void kernel_launch(void* const* d_in, const int* in_sizes, int n_in,
                              void* d_out, int out_size, void* d_ws, size_t ws_size,
                              hipStream_t stream) {
  const float* state = (const float*)d_in[0];   // [1024,256]
  const float* W     = (const float*)d_in[1];   // [256,128]
  const float* bias  = (const float*)d_in[2];   // [128]
  const float* items = (const float*)d_in[3];   // [100000,128]

  char* ws = (char*)d_ws;
  size_t off = 0;
  unsigned short* actBf16   = (unsigned short*)(ws + off); off += (size_t)B_ * D_ * 2;
  unsigned short* itemsBf16 = (unsigned short*)(ws + off); off += (size_t)N_ * D_ * 2;
  off = (off + 255) & ~(size_t)255;
  float* pscore = (float*)(ws + off); off += (size_t)B_ * NCHUNK * K_ * 4;
  off = (off + 255) & ~(size_t)255;
  int* pidx = (int*)(ws + off);
  int* out = (int*)d_out;                       // int32 indices [1024,10]

  actor_norm_kernel<<<B_, D_, 0, stream>>>(state, W, bias, actBf16);
  item_norm_kernel<<<N_ / 8, 256, 0, stream>>>(items, itemsBf16);
  dim3 g3(NCHUNK, B_ / ROWS_PER_BLOCK);
  score_topk_kernel<<<g3, 256, 0, stream>>>(actBf16, itemsBf16, pscore, pidx);
  merge_topk_kernel<<<B_, 32, 0, stream>>>(pscore, pidx, out);
}
